// GPT_88656714925257
// MI455X (gfx1250) — compile-verified, matrix operand order checked by reading, not run on hardware
//
#include <hip/hip_runtime.h>

// ---------------------------------------------------------------------------
// GPT-2 small forward on gfx1250 (CDNA5, wave32, WMMA bf16)
// B=4, T=1024, L=12, C=768, H=12, D=64, F4=3072, V=50257
// ---------------------------------------------------------------------------

typedef __attribute__((ext_vector_type(16))) __bf16 bf16x16;
typedef __attribute__((ext_vector_type(8)))  float  f32x8;

union AFrag { bf16x16 v; unsigned u[8]; __bf16 e[16]; };
union CAcc  { f32x8   v; float    f[8]; };
union Pack4 { __bf16 h[4]; uint2 u2; };

// k index (bf16 element, always even) held in VGPR j of a 16-bit A/B fragment
// (16x32, K=32): lanes 0-15 -> K {0..7,16..23}, lanes 16-31 -> +8.
__device__ __forceinline__ int kmap(int j, int hf) {
  return ((j < 4) ? (2 * j) : (16 + 2 * (j - 4))) + 8 * hf;
}

__device__ __forceinline__ f32x8 wmma_bf16(bf16x16 a, bf16x16 b, f32x8 c) {
  return __builtin_amdgcn_wmma_f32_16x16x32_bf16(false, a, false, b,
                                                 (short)0, c, false, false);
}

__device__ __forceinline__ float gelu_tanh(float x) {
  const float k0 = 0.7978845608028654f;  // sqrt(2/pi)
  const float k1 = 0.044715f;
  return 0.5f * x * (1.f + tanhf(k0 * (x + k1 * x * x * x)));
}

// ---------------------------------------------------------------------------
// Embedding
// ---------------------------------------------------------------------------
__global__ __launch_bounds__(256) void embed_k(const int* __restrict__ x,
                                               const float* __restrict__ wte,
                                               const float* __restrict__ wpe,
                                               float* __restrict__ h) {
  const int tok = blockIdx.x;
  const int t   = tok & 1023;
  const int id  = x[tok];
  for (int i = threadIdx.x; i < 768; i += 256)
    h[(size_t)tok * 768 + i] = wte[(size_t)id * 768 + i] + wpe[(size_t)t * 768 + i];
}

// ---------------------------------------------------------------------------
// LayerNorm over C=768, one block per token row
// ---------------------------------------------------------------------------
__global__ __launch_bounds__(256) void layernorm_k(const float* __restrict__ x,
                                                   const float* __restrict__ w,
                                                   const float* __restrict__ b,
                                                   float* __restrict__ o) {
  const int row = blockIdx.x;
  const float* xr = x + (size_t)row * 768;
  const int c = threadIdx.x;
  float v0 = xr[c], v1 = xr[c + 256], v2 = xr[c + 512];
  float s = v0 + v1 + v2;
  float s2 = v0 * v0 + v1 * v1 + v2 * v2;
#pragma unroll
  for (int off = 16; off > 0; off >>= 1) {
    s  += __shfl_xor(s,  off, 32);
    s2 += __shfl_xor(s2, off, 32);
  }
  __shared__ float ss[8], ss2[8];
  if ((c & 31) == 0) { ss[c >> 5] = s; ss2[c >> 5] = s2; }
  __syncthreads();
  s = 0.f; s2 = 0.f;
#pragma unroll
  for (int i = 0; i < 8; ++i) { s += ss[i]; s2 += ss2[i]; }
  const float mu  = s * (1.f / 768.f);
  const float var = s2 * (1.f / 768.f) - mu * mu;
  const float r   = rsqrtf(var + 1e-5f);
  float* orow = o + (size_t)row * 768;
  orow[c]       = (v0 - mu) * r * w[c]       + b[c];
  orow[c + 256] = (v1 - mu) * r * w[c + 256] + b[c + 256];
  orow[c + 512] = (v2 - mu) * r * w[c + 512] + b[c + 512];
}

// ---------------------------------------------------------------------------
// WMMA GEMM: Cout[M,N] = act(A[M,K] @ W[K,N] + bias) + resid
// Block tile 128x256xBK64, 8 waves (2x4), wave tile 64x64 (4x4 WMMA accums).
// Register-pipelined: next stage's global loads issued before current compute.
// K must be a multiple of 64 (768 / 3072 both are).
// ---------------------------------------------------------------------------
__global__ __launch_bounds__(256) void gemm_bf16_wmma(
    const float* __restrict__ A, const float* __restrict__ W,
    const float* __restrict__ bias, const float* __restrict__ resid,
    float* __restrict__ Cout, int M, int N, int K, int act) {
  __shared__ __align__(16) __bf16 As[128][64];    // 16 KB   [m][k]
  __shared__ __align__(16) __bf16 BsT[256][64];   // 32 KB   [n][k]
  const int tid  = threadIdx.x;
  const int lane = tid & 31, wave = tid >> 5;
  const int wm = wave >> 2, wn = wave & 3;        // 2 x 4 waves -> 128 x 256
  const int lm = lane & 15, hf = lane >> 4;
  const int bm = blockIdx.y * 128, bn = blockIdx.x * 256;

  // ---- per-thread staging descriptors (precomputed base pointers) ----
  // A tile: 128x64 = 2048 float4, 8 per thread
  const float* aPtr[8];
  int aR[8], aC[8];
#pragma unroll
  for (int i = 0; i < 8; ++i) {
    int f = i * 256 + tid;
    aR[i] = f >> 4;            // row 0..127
    aC[i] = (f & 15) * 4;      // k offset 0..60
    aPtr[i] = A + (size_t)(bm + aR[i]) * K + aC[i];
  }
  // B tile: 64x256 floats = 16 quads of 4 scalars per thread
  const float* bPtr[16];
  int bN[16], bK[16];
#pragma unroll
  for (int i = 0; i < 16; ++i) {
    int f = i * 256 + tid;
    bK[i] = f >> 6;            // k row 0..63
    bN[i] = (f & 63) * 4;      // n offset 0..252
    bPtr[i] = W + (size_t)bK[i] * N + (bn + bN[i]);
  }

  CAcc acc[4][4];
#pragma unroll
  for (int mi = 0; mi < 4; ++mi)
#pragma unroll
    for (int ni = 0; ni < 4; ++ni)
#pragma unroll
      for (int j = 0; j < 8; ++j) acc[mi][ni].f[j] = 0.f;

  float4 aReg[8];
  float  bReg[16][4];

  // ---- prologue: load + stage tile 0 ----
#pragma unroll
  for (int i = 0; i < 8; ++i) aReg[i] = *(const float4*)aPtr[i];
#pragma unroll
  for (int i = 0; i < 16; ++i)
#pragma unroll
    for (int t = 0; t < 4; ++t) {
      int gn = bn + bN[i] + t;
      bReg[i][t] = (gn < N) ? bPtr[i][t] : 0.f;
    }
#pragma unroll
  for (int i = 0; i < 8; ++i) {
    Pack4 p;
    p.h[0] = (__bf16)aReg[i].x; p.h[1] = (__bf16)aReg[i].y;
    p.h[2] = (__bf16)aReg[i].z; p.h[3] = (__bf16)aReg[i].w;
    *(uint2*)&As[aR[i]][aC[i]] = p.u2;
  }
#pragma unroll
  for (int i = 0; i < 16; ++i)
#pragma unroll
    for (int t = 0; t < 4; ++t)
      BsT[bN[i] + t][bK[i]] = (__bf16)bReg[i][t];
  __syncthreads();

  for (int k0 = 0; k0 < K; k0 += 64) {
    const int nk = k0 + 64;
    const bool more = nk < K;
    // issue next stage's global loads early (overlap with WMMA below)
    if (more) {
#pragma unroll
      for (int i = 0; i < 8; ++i) aReg[i] = *(const float4*)(aPtr[i] + nk);
#pragma unroll
      for (int i = 0; i < 16; ++i) {
        const float* p = bPtr[i] + (size_t)nk * N;
#pragma unroll
        for (int t = 0; t < 4; ++t) {
          int gn = bn + bN[i] + t;
          bReg[i][t] = (gn < N) ? p[t] : 0.f;
        }
      }
    }
    // compute current stage: 2 k-chunks x 16 WMMA
#pragma unroll
    for (int c = 0; c < 2; ++c) {
      AFrag af[4], bf_[4];
#pragma unroll
      for (int mi = 0; mi < 4; ++mi) {
        const __bf16* p = &As[wm * 64 + mi * 16 + lm][c * 32];
#pragma unroll
        for (int j = 0; j < 8; ++j)
          af[mi].u[j] = *(const unsigned*)(p + kmap(j, hf));
      }
#pragma unroll
      for (int ni = 0; ni < 4; ++ni) {
        const __bf16* p = &BsT[wn * 64 + ni * 16 + lm][c * 32];
#pragma unroll
        for (int j = 0; j < 8; ++j)
          bf_[ni].u[j] = *(const unsigned*)(p + kmap(j, hf));
      }
#pragma unroll
      for (int mi = 0; mi < 4; ++mi)
#pragma unroll
        for (int ni = 0; ni < 4; ++ni)
          acc[mi][ni].v = wmma_bf16(af[mi].v, bf_[ni].v, acc[mi][ni].v);
    }
    if (more) {
      __syncthreads();  // all reads of LDS done before overwrite
#pragma unroll
      for (int i = 0; i < 8; ++i) {
        Pack4 p;
        p.h[0] = (__bf16)aReg[i].x; p.h[1] = (__bf16)aReg[i].y;
        p.h[2] = (__bf16)aReg[i].z; p.h[3] = (__bf16)aReg[i].w;
        *(uint2*)&As[aR[i]][aC[i]] = p.u2;
      }
#pragma unroll
      for (int i = 0; i < 16; ++i)
#pragma unroll
        for (int t = 0; t < 4; ++t)
          BsT[bN[i] + t][bK[i]] = (__bf16)bReg[i][t];
      __syncthreads();
    }
  }

  // epilogue
#pragma unroll
  for (int mi = 0; mi < 4; ++mi)
#pragma unroll
    for (int ni = 0; ni < 4; ++ni)
#pragma unroll
      for (int j = 0; j < 8; ++j) {
        int row = bm + wm * 64 + mi * 16 + j + 8 * hf;
        int col = bn + wn * 64 + ni * 16 + lm;
        if (row < M && col < N) {
          float v = acc[mi][ni].f[j];
          if (bias)     v += bias[col];
          if (act == 1) v  = gelu_tanh(v);
          if (resid)    v += resid[(size_t)row * N + col];
          Cout[(size_t)row * N + col] = v;
        }
      }
}

// ---------------------------------------------------------------------------
// Flash-style causal attention. qkv: [B*T, 3C]; y: [B*T, C].
// grid = (B*H, T/128); block = 256 (8 waves; wave w owns 16 q rows)
// ---------------------------------------------------------------------------
__global__ __launch_bounds__(256) void attn_wmma(const float* __restrict__ qkv,
                                                 float* __restrict__ y) {
  __shared__ __align__(16) __bf16 Kt[32][64];       // [s][d]
  __shared__ __align__(16) __bf16 Vt[64][32];       // [d][s] (transposed)
  __shared__ __align__(16) __bf16 Pt[8][16][32];    // per-wave P scratch
  const int bh = blockIdx.x;
  const int hh = bh % 12, bb = bh / 12;
  const int qtile = blockIdx.y;
  const int tid = threadIdx.x, lane = tid & 31, wave = tid >> 5;
  const int lm = lane & 15, hf = lane >> 4;
  const float* base = qkv + (size_t)bb * 1024 * 2304;
  const int q0 = qtile * 128 + wave * 16;

  // staging descriptors: K/V tile 32x64 = 512 float4 each, 2 per thread
  const float* kPtr[2];
  int sA[2], dA[2];
#pragma unroll
  for (int i = 0; i < 2; ++i) {
    int f = i * 256 + tid;
    sA[i] = f >> 4;
    dA[i] = (f & 15) * 4;
    kPtr[i] = base + (size_t)sA[i] * 2304 + 768 + hh * 64 + dA[i];
  }

  // Q fragments (A layout), pre-scaled by 1/sqrt(D); batched float2 loads
  AFrag qf[2];
  {
    const float* qrow = base + (size_t)(q0 + lm) * 2304 + hh * 64;
    float2 qr[2][8];
#pragma unroll
    for (int c = 0; c < 2; ++c)
#pragma unroll
      for (int j = 0; j < 8; ++j)
        qr[c][j] = *(const float2*)(qrow + c * 32 + kmap(j, hf));
#pragma unroll
    for (int c = 0; c < 2; ++c)
#pragma unroll
      for (int j = 0; j < 8; ++j) {
        qf[c].e[2 * j]     = (__bf16)(qr[c][j].x * 0.125f);
        qf[c].e[2 * j + 1] = (__bf16)(qr[c][j].y * 0.125f);
      }
  }

  CAcc o[4];
#pragma unroll
  for (int nb = 0; nb < 4; ++nb)
#pragma unroll
    for (int j = 0; j < 8; ++j) o[nb].f[j] = 0.f;
  float mrow[8], lrow[8];
#pragma unroll
  for (int j = 0; j < 8; ++j) { mrow[j] = -3.0e38f; lrow[j] = 0.f; }

  const int kv_end = qtile * 128 + 128;
  for (int s0 = 0; s0 < kv_end; s0 += 32) {
    // cooperative K/V tile stage, batched float4 loads
    float4 kReg[2], vReg[2];
#pragma unroll
    for (int i = 0; i < 2; ++i) {
      const float* kp = kPtr[i] + (size_t)s0 * 2304;
      kReg[i] = *(const float4*)kp;
      vReg[i] = *(const float4*)(kp + 768);
    }
#pragma unroll
    for (int i = 0; i < 2; ++i) {
      Pack4 p;
      p.h[0] = (__bf16)kReg[i].x; p.h[1] = (__bf16)kReg[i].y;
      p.h[2] = (__bf16)kReg[i].z; p.h[3] = (__bf16)kReg[i].w;
      *(uint2*)&Kt[sA[i]][dA[i]] = p.u2;
      Vt[dA[i] + 0][sA[i]] = (__bf16)vReg[i].x;
      Vt[dA[i] + 1][sA[i]] = (__bf16)vReg[i].y;
      Vt[dA[i] + 2][sA[i]] = (__bf16)vReg[i].z;
      Vt[dA[i] + 3][sA[i]] = (__bf16)vReg[i].w;
    }
    __syncthreads();

    if (s0 <= q0 + 15) {  // wave-uniform -> EXEC all-1s around WMMA
      CAcc S[2];
#pragma unroll
      for (int t = 0; t < 2; ++t) {
#pragma unroll
        for (int j = 0; j < 8; ++j) S[t].f[j] = 0.f;
#pragma unroll
        for (int c = 0; c < 2; ++c) {
          AFrag kf;
          const __bf16* kp = &Kt[t * 16 + lm][c * 32];
#pragma unroll
          for (int j = 0; j < 8; ++j)
            kf.u[j] = *(const unsigned*)(kp + kmap(j, hf));
          S[t].v = wmma_bf16(qf[c].v, kf.v, S[t].v);
        }
      }

      // causal mask + online softmax
      float rescale[8];
#pragma unroll
      for (int j = 0; j < 8; ++j) {
        const int q = q0 + j + 8 * hf;
#pragma unroll
        for (int t = 0; t < 2; ++t) {
          int s = s0 + t * 16 + lm;
          if (s > q) S[t].f[j] = -1.0e30f;
        }
        float v = fmaxf(S[0].f[j], S[1].f[j]);
        v = fmaxf(v, __shfl_xor(v, 1, 32));
        v = fmaxf(v, __shfl_xor(v, 2, 32));
        v = fmaxf(v, __shfl_xor(v, 4, 32));
        v = fmaxf(v, __shfl_xor(v, 8, 32));
        const float mnew = fmaxf(mrow[j], v);
        rescale[j] = __expf(mrow[j] - mnew);
        const float p0 = __expf(S[0].f[j] - mnew);
        const float p1 = __expf(S[1].f[j] - mnew);
        float rs = p0 + p1;
        rs += __shfl_xor(rs, 1, 32);
        rs += __shfl_xor(rs, 2, 32);
        rs += __shfl_xor(rs, 4, 32);
        rs += __shfl_xor(rs, 8, 32);
        lrow[j] = lrow[j] * rescale[j] + rs;
        mrow[j] = mnew;
        Pt[wave][j + 8 * hf][lm]      = (__bf16)p0;
        Pt[wave][j + 8 * hf][16 + lm] = (__bf16)p1;
      }
      asm volatile("s_wait_dscnt 0" ::: "memory");  // P store -> reload (same wave)

      AFrag pf;
      {
        const __bf16* pp = &Pt[wave][lm][0];
#pragma unroll
        for (int j = 0; j < 8; ++j)
          pf.u[j] = *(const unsigned*)(pp + kmap(j, hf));
      }
#pragma unroll
      for (int nb = 0; nb < 4; ++nb) {
#pragma unroll
        for (int j = 0; j < 8; ++j) o[nb].f[j] *= rescale[j];
        AFrag vf;
        const __bf16* vp = &Vt[nb * 16 + lm][0];
#pragma unroll
        for (int j = 0; j < 8; ++j)
          vf.u[j] = *(const unsigned*)(vp + kmap(j, hf));
        o[nb].v = wmma_bf16(pf.v, vf.v, o[nb].v);
      }
    }
    __syncthreads();
  }

  float* yb = y + (size_t)bb * 1024 * 768;
#pragma unroll
  for (int nb = 0; nb < 4; ++nb)
#pragma unroll
    for (int j = 0; j < 8; ++j) {
      int q = q0 + j + 8 * hf;
      int d = hh * 64 + nb * 16 + lm;
      yb[(size_t)q * 768 + d] = o[nb].f[j] / lrow[j];
    }
}

// ---------------------------------------------------------------------------
// Host-side orchestration
// ---------------------------------------------------------------------------
extern "C" void kernel_launch(void* const* d_in, const int* in_sizes, int n_in,
                              void* d_out, int out_size, void* d_ws, size_t ws_size,
                              hipStream_t stream) {
  (void)in_sizes; (void)n_in; (void)out_size; (void)ws_size;
  const int*   x       = (const int*)  d_in[0];
  const float* wte     = (const float*)d_in[1];
  const float* wpe     = (const float*)d_in[2];
  const float* ln1_w   = (const float*)d_in[3];
  const float* ln1_b   = (const float*)d_in[4];
  const float* attn_w  = (const float*)d_in[5];
  const float* attn_b  = (const float*)d_in[6];
  const float* attnp_w = (const float*)d_in[7];
  const float* attnp_b = (const float*)d_in[8];
  const float* ln2_w   = (const float*)d_in[9];
  const float* ln2_b   = (const float*)d_in[10];
  const float* fc_w    = (const float*)d_in[11];
  const float* fc_b    = (const float*)d_in[12];
  const float* mproj_w = (const float*)d_in[13];
  const float* mproj_b = (const float*)d_in[14];
  const float* lnf_w   = (const float*)d_in[15];
  const float* lnf_b   = (const float*)d_in[16];
  const float* lm_w    = (const float*)d_in[17];
  float* out = (float*)d_out;

  const int M = 4096;  // B*T tokens
  float* h   = (float*)d_ws;            // [4096, 768]
  float* hn  = h   + (size_t)M * 768;   // [4096, 768]
  float* qkv = hn  + (size_t)M * 768;   // [4096, 2304]
  float* yb  = qkv + (size_t)M * 2304;  // [4096, 768]
  float* hm  = yb  + (size_t)M * 768;   // [4096, 3072]

  const dim3 thr(256);
  embed_k<<<M, thr, 0, stream>>>(x, wte, wpe, h);

  for (int l = 0; l < 12; ++l) {
    layernorm_k<<<M, thr, 0, stream>>>(h, ln1_w + l * 768, ln1_b + l * 768, hn);
    gemm_bf16_wmma<<<dim3(9, 32), thr, 0, stream>>>(
        hn, attn_w + (size_t)l * 768 * 2304, attn_b + (size_t)l * 2304,
        nullptr, qkv, M, 2304, 768, 0);
    attn_wmma<<<dim3(48, 8), thr, 0, stream>>>(qkv, yb);
    gemm_bf16_wmma<<<dim3(3, 32), thr, 0, stream>>>(
        yb, attnp_w + (size_t)l * 768 * 768, attnp_b + (size_t)l * 768,
        h, h, M, 768, 768, 0);
    layernorm_k<<<M, thr, 0, stream>>>(h, ln2_w + l * 768, ln2_b + l * 768, hn);
    gemm_bf16_wmma<<<dim3(12, 32), thr, 0, stream>>>(
        hn, fc_w + (size_t)l * 768 * 3072, fc_b + (size_t)l * 3072,
        nullptr, hm, M, 3072, 768, 1);
    gemm_bf16_wmma<<<dim3(3, 32), thr, 0, stream>>>(
        hm, mproj_w + (size_t)l * 3072 * 768, mproj_b + (size_t)l * 768,
        h, h, M, 768, 3072, 0);
  }

  layernorm_k<<<M, thr, 0, stream>>>(h, lnf_w, lnf_b, hn);
  gemm_bf16_wmma<<<dim3(197, 32), thr, 0, stream>>>(
      hn, lm_w, nullptr, nullptr, out, M, 50257, 768, 0);
}
